// RetinaNet_1563368096201
// MI455X (gfx1250) — matrix operand order repaired and checked by reading, not compiled, not run on hardware
//
#include <hip/hip_runtime.h>
#include <hip/hip_bf16.h>

typedef __attribute__((ext_vector_type(16))) __bf16 v16bf;
typedef __attribute__((ext_vector_type(8)))  __bf16 v8bf;
typedef __attribute__((ext_vector_type(8)))  float  v8f;

// ---------------- helpers ----------------

__device__ __forceinline__ __bf16 f2bf(float f) {
    unsigned u = __builtin_bit_cast(unsigned, f);
    unsigned r = u + 0x7FFFu + ((u >> 16) & 1u);   // round-to-nearest-even
    unsigned short h = (unsigned short)(r >> 16);
    return __builtin_bit_cast(__bf16, h);
}

// combine two 8-element bf16 chunks (each 16B aligned) into one v16bf fragment
__device__ __forceinline__ v16bf combine16(const __bf16* p0, const __bf16* p1) {
    v8bf lo = *(const v8bf*)p0;
    v8bf hi = *(const v8bf*)p1;
    v16bf r;
#pragma unroll
    for (int e = 0; e < 8; ++e) { r[e] = lo[e]; r[e + 8] = hi[e]; }
    return r;
}

// ---------------- weight packing: f32 [Cout][256][3][3] -> bf16 [Cout][2304], k = tap*256+ci ----------------

__global__ void pack_w_kernel(const float* __restrict__ src, __bf16* __restrict__ dst,
                              int rows, int rowOff) {
    int idx = blockIdx.x * 256 + threadIdx.x;
    if (idx >= rows * 2304) return;
    int co = idx / 2304, k = idx % 2304;
    int t = k >> 8, ci = k & 255;
    dst[(size_t)(rowOff + co) * 2304 + k] = f2bf(src[((size_t)co * 256 + ci) * 9 + t]);
}

// ---------------- activation convert: f32 NCHW -> bf16 NHWC (C=256) ----------------

__global__ void cvt_kernel(const float* __restrict__ src, __bf16* __restrict__ dst,
                           int Bn, int HW) {
    int idx = blockIdx.x * 256 + threadIdx.x;
    if (idx >= Bn * HW * 256) return;
    int c = idx & 255;
    int rest = idx >> 8;
    int p = rest % HW;
    int b = rest / HW;
    dst[idx] = f2bf(src[((size_t)b * 256 + c) * HW + p]);
}

// ---------------- 3x3 SAME conv as implicit GEMM via WMMA bf16 ----------------
// X: bf16 NHWC (Cin=256), Wp: bf16 [Cout][2304], Y: f32 NHWC (Cout) with bias added.
// Block: 256 thr = 8 waves; block tile 64 Cout x 128 pixels.
// Wave tile 32 Cout x 32 pixels = 4 WMMA accumulators (2 A-frags x 2 B-frags).
// K loop: 9 taps x 8 chunks of 32; tap address math hoisted; global loads pipelined.

__global__ __launch_bounds__(256) void conv3x3_wmma_kernel(
    const __bf16* __restrict__ X, const __bf16* __restrict__ Wp,
    const float* __restrict__ bias, float* __restrict__ Y,
    int Bn, int H, int W, int Cout)
{
    const int HW   = H * W;
    const int Ntot = Bn * HW;
    const int tid  = threadIdx.x;
    const int lane = tid & 31;
    const int wv   = tid >> 5;
    const int wm   = wv & 1;     // Cout half (32 rows)
    const int wn   = wv >> 1;    // pixel quarter (32 cols)
    const int nblk  = blockIdx.x * 128;
    const int coblk = blockIdx.y * 64;

    __shared__ __bf16 As[64][32];    // [co_local][k_local]    4 KB
    __shared__ __bf16 Bs[128][32];   // [pixel_local][k_local] 8 KB

    // ---- staging coordinates (fixed per thread) ----
    // B: each thread owns pixel ip, 16-wide ci half  -> 2 x b128 per chunk
    const int ip   = tid >> 1;        // 0..127
    const int bcig = (tid & 1) * 16;  // 0 or 16
    const int N = nblk + ip;
    const bool nvalid = N < Ntot;
    int b = 0, h = 0, w = 0;
    if (nvalid) { b = N / HW; int p = N % HW; h = p / W; w = p % W; }

    // A: each thread owns co row arow, 8-wide k sub-chunk -> 1 x b128 per chunk
    const int arow = tid >> 2;        // 0..63
    const int akg  = (tid & 3) * 8;
    const int aco  = coblk + arow;
    const bool avalid = aco < Cout;
    const __bf16* wrow = Wp + (size_t)aco * 2304;

    v8f c00 = {0.f,0.f,0.f,0.f,0.f,0.f,0.f,0.f};
    v8f c01 = c00, c10 = c00, c11 = c00;

    const int m  = lane & 15;
    const int hl = lane >> 4;

    for (int tap = 0; tap < 9; ++tap) {
        const int kh = tap / 3;
        const int kw = tap - kh * 3;
        const int hh = h + kh - 1;
        const int ww = w + kw - 1;
        const bool bval = nvalid && hh >= 0 && hh < H && ww >= 0 && ww < W;
        const __bf16* bsrc = X + ((size_t)b * HW + (size_t)(hh * W + ww)) * 256 + bcig;
        const __bf16* asrc = wrow + tap * 256 + akg;
        if (avalid && tap < 8) __builtin_prefetch(wrow + (tap + 1) * 256 + akg, 0, 1);

        uint4 av = {0u,0u,0u,0u}, bv0 = {0u,0u,0u,0u}, bv1 = {0u,0u,0u,0u};
        if (avalid) av  = *(const uint4*)(asrc);
        if (bval)   { bv0 = *(const uint4*)(bsrc); bv1 = *(const uint4*)(bsrc + 8); }

        for (int kc = 0; kc < 256; kc += 32) {
            __syncthreads();
            *(uint4*)&As[arow][akg]     = av;
            *(uint4*)&Bs[ip][bcig]      = bv0;
            *(uint4*)&Bs[ip][bcig + 8]  = bv1;
            __syncthreads();

            // pipeline: issue next chunk's global loads before LDS reads
            uint4 avn = {0u,0u,0u,0u}, bv0n = {0u,0u,0u,0u}, bv1n = {0u,0u,0u,0u};
            if (kc < 224) {
                if (avalid) avn  = *(const uint4*)(asrc + kc + 32);
                if (bval)   { bv0n = *(const uint4*)(bsrc + kc + 32);
                              bv1n = *(const uint4*)(bsrc + kc + 40); }
            }

            // fragments (contiguous k order, same for A and B):
            // A row=co, K(e)=e+16*hl ; B col=px, K(e)=e+16*hl
            v16bf a0 = combine16(&As[wm*32 + m][16*hl],      &As[wm*32 + m][16*hl + 8]);
            v16bf a1 = combine16(&As[wm*32 + 16 + m][16*hl], &As[wm*32 + 16 + m][16*hl + 8]);
            v16bf p0 = combine16(&Bs[wn*32 + m][16*hl],      &Bs[wn*32 + m][16*hl + 8]);
            v16bf p1 = combine16(&Bs[wn*32 + 16 + m][16*hl], &Bs[wn*32 + 16 + m][16*hl + 8]);

            c00 = __builtin_amdgcn_wmma_f32_16x16x32_bf16(false, a0, false, p0, (short)0, c00, false, false);
            c01 = __builtin_amdgcn_wmma_f32_16x16x32_bf16(false, a0, false, p1, (short)0, c01, false, false);
            c10 = __builtin_amdgcn_wmma_f32_16x16x32_bf16(false, a1, false, p0, (short)0, c10, false, false);
            c11 = __builtin_amdgcn_wmma_f32_16x16x32_bf16(false, a1, false, p1, (short)0, c11, false, false);

            av = avn; bv0 = bv0n; bv1 = bv1n;
        }
    }

    // C layout: lane L: col N = L&15, rows M = r + 8*(L>>4)
    const int ncol = lane & 15;
#pragma unroll
    for (int fa = 0; fa < 2; ++fa) {
        const v8f* ca = fa ? &c10 : &c00;
        const v8f* cb = fa ? &c11 : &c01;
#pragma unroll
        for (int r = 0; r < 8; ++r) {
            int co = coblk + wm*32 + fa*16 + r + 8*hl;
            if (co < Cout) {
                float bco = bias[co];
                int n0 = nblk + wn*32 + ncol;
                if (n0 < Ntot) Y[(size_t)n0 * Cout + co] = (*ca)[r] + bco;
                int n1 = n0 + 16;
                if (n1 < Ntot) Y[(size_t)n1 * Cout + co] = (*cb)[r] + bco;
            }
        }
    }
}

// ---------------- GroupNorm(32 groups over C=256) + ReLU: f32 NHWC -> bf16 NHWC ----------------

__global__ __launch_bounds__(256) void gn_relu_kernel(
    const float* __restrict__ Yin, const float* __restrict__ gamma,
    const float* __restrict__ beta, __bf16* __restrict__ Xout, int HW)
{
    const int b  = blockIdx.x >> 5;
    const int g  = blockIdx.x & 31;
    const int cg = g * 8;
    const int tid = threadIdx.x;
    const int cnt = HW * 8;
    const float* base = Yin + (size_t)b * HW * 256;

    float s = 0.f, ss = 0.f;
    for (int i = tid; i < cnt; i += 256) {
        int p = i >> 3, c = i & 7;
        float v = base[(size_t)p * 256 + cg + c];
        s += v; ss += v * v;
    }
    __shared__ float r1[256], r2[256];
    r1[tid] = s; r2[tid] = ss;
    __syncthreads();
    for (int st = 128; st > 0; st >>= 1) {
        if (tid < st) { r1[tid] += r1[tid + st]; r2[tid] += r2[tid + st]; }
        __syncthreads();
    }
    float inv  = 1.f / (float)cnt;
    float mean = r1[0] * inv;
    float var  = r2[0] * inv - mean * mean;
    float rstd = rsqrtf(var + 1e-5f);

    __bf16* obase = Xout + (size_t)b * HW * 256;
    for (int i = tid; i < cnt; i += 256) {
        int p = i >> 3, c = i & 7;
        float v = base[(size_t)p * 256 + cg + c];
        float y = (v - mean) * rstd * gamma[cg + c] + beta[cg + c];
        obase[(size_t)p * 256 + cg + c] = f2bf(fmaxf(y, 0.f));
    }
}

// ---------------- tiny fixups ----------------

__global__ void bias5_kernel(const float* regb, const float* ioub, float* b5) {
    int t = threadIdx.x;
    if (t < 4) b5[t] = regb[t];
    else if (t == 4) b5[4] = ioub[0];
}

__global__ void copy_cls_kernel(const float* __restrict__ cls, float* __restrict__ out,
                                int Ntot, int HW, int off) {
    int idx = blockIdx.x * 256 + threadIdx.x;
    if (idx >= Ntot * 80) return;
    int n = idx / 80, c = idx % 80;
    int b = n / HW, p = n % HW;
    out[((size_t)b * 8525 + off + p) * 85 + 5 + c] = cls[idx];
}

__global__ void decode_kernel(const float* __restrict__ rio, const float* __restrict__ scales,
                              int lvl, float* __restrict__ out,
                              int Ntot, int HW, int W, float stride, float asize, int off) {
    int n = blockIdx.x * 256 + threadIdx.x;
    if (n >= Ntot) return;
    int b = n / HW, p = n % HW;
    int h = p / W, w = p % W;
    float sc = scales[lvl];
    const float* t = rio + (size_t)n * 5;
    float r0 = t[0] * sc, r1 = t[1] * sc, r2 = t[2] * sc, r3 = t[3] * sc, iou = t[4];
    float cx = (w + 0.5f) * stride, cy = (h + 0.5f) * stride;
    float x = r0 * asize + cx, y = r1 * asize + cy;
    float bw = expf(r2) * asize, bh = expf(r3) * asize;
    size_t base = ((size_t)b * 8525 + off + p) * 85;
    out[base + 0] = x - 0.5f * bw;
    out[base + 1] = y - 0.5f * bh;
    out[base + 2] = x + 0.5f * bw;
    out[base + 3] = y + 0.5f * bh;
    out[base + 4] = iou;
}

// ---------------- host launcher ----------------

extern "C" void kernel_launch(void* const* d_in, const int* in_sizes, int n_in,
                              void* d_out, int out_size, void* d_ws, size_t ws_size,
                              hipStream_t stream) {
    (void)in_sizes; (void)n_in; (void)out_size; (void)ws_size;

    const float* feats[5];
    for (int i = 0; i < 5; ++i) feats[i] = (const float*)d_in[i];
    const float* cls_tw    = (const float*)d_in[5];
    const float* cls_tb    = (const float*)d_in[6];
    const float* cls_tg    = (const float*)d_in[7];
    const float* cls_tbeta = (const float*)d_in[8];
    const float* reg_tw    = (const float*)d_in[9];
    const float* reg_tb    = (const float*)d_in[10];
    const float* reg_tg    = (const float*)d_in[11];
    const float* reg_tbeta = (const float*)d_in[12];
    const float* cls_hw    = (const float*)d_in[13];
    const float* cls_hb    = (const float*)d_in[14];
    const float* reg_hw    = (const float*)d_in[15];
    const float* reg_hb    = (const float*)d_in[16];
    const float* iou_hw    = (const float*)d_in[17];
    const float* iou_hb    = (const float*)d_in[18];
    const float* scales    = (const float*)d_in[19];
    float* out = (float*)d_out;

    const int Bn = 8;
    static const int   Hs[5]      = {80, 40, 20, 10, 5};
    static const float strides[5] = {8.f, 16.f, 32.f, 64.f, 128.f};
    static const float asz[5]     = {32.f, 64.f, 128.f, 256.f, 512.f};
    static const int   offs[5]    = {0, 6400, 8000, 8400, 8500};

    // workspace carve-up (all 256B aligned)
    char* ws = (char*)d_ws;
    size_t cur = 0;
    auto carve = [&](size_t bytes) -> char* {
        char* p = ws + cur;
        cur = (cur + bytes + 255) & ~(size_t)255;
        return p;
    };
    const size_t maxHW = 80 * 80;
    __bf16* wpA  = (__bf16*)carve((size_t)4 * 256 * 2304 * sizeof(__bf16)); // cls tower
    __bf16* wpB  = (__bf16*)carve((size_t)4 * 256 * 2304 * sizeof(__bf16)); // reg tower
    __bf16* wpCH = (__bf16*)carve((size_t)80 * 2304 * sizeof(__bf16));      // cls head
    __bf16* wpRI = (__bf16*)carve((size_t)5 * 2304 * sizeof(__bf16));       // reg(4)+iou(1) head
    float*  b5   = (float*)carve(5 * sizeof(float));
    __bf16* xbf  = (__bf16*)carve((size_t)Bn * maxHW * 256 * sizeof(__bf16)); // level input, bf16 NHWC
    __bf16* abf  = (__bf16*)carve((size_t)Bn * maxHW * 256 * sizeof(__bf16)); // tower activation
    float*  tmp  = (float*)carve((size_t)Bn * maxHW * 256 * sizeof(float));   // conv outputs f32 NHWC

    // ---- pack weights ----
    {
        int g256 = (256 * 2304 + 255) / 256;
        for (int i = 0; i < 4; ++i) {
            pack_w_kernel<<<g256, 256, 0, stream>>>(cls_tw + (size_t)i * 256 * 256 * 9,
                                                    wpA + (size_t)i * 256 * 2304, 256, 0);
            pack_w_kernel<<<g256, 256, 0, stream>>>(reg_tw + (size_t)i * 256 * 256 * 9,
                                                    wpB + (size_t)i * 256 * 2304, 256, 0);
        }
        pack_w_kernel<<<(80 * 2304 + 255) / 256, 256, 0, stream>>>(cls_hw, wpCH, 80, 0);
        pack_w_kernel<<<(4 * 2304 + 255) / 256, 256, 0, stream>>>(reg_hw, wpRI, 4, 0);
        pack_w_kernel<<<(1 * 2304 + 255) / 256, 256, 0, stream>>>(iou_hw, wpRI, 1, 4);
        bias5_kernel<<<1, 32, 0, stream>>>(reg_hb, iou_hb, b5);
    }

    for (int j = 0; j < 5; ++j) {
        const int H = Hs[j], W = Hs[j], HW = H * W;
        const int Ntot = Bn * HW;
        const int nblks = (Ntot + 127) / 128;

        cvt_kernel<<<(Bn * HW * 256 + 255) / 256, 256, 0, stream>>>(feats[j], xbf, Bn, HW);

        // ---- cls tower ----
        const __bf16* curx = xbf;
        for (int i = 0; i < 4; ++i) {
            dim3 grid(nblks, 4);
            conv3x3_wmma_kernel<<<grid, 256, 0, stream>>>(curx, wpA + (size_t)i * 256 * 2304,
                                                          cls_tb + i * 256, tmp, Bn, H, W, 256);
            gn_relu_kernel<<<Bn * 32, 256, 0, stream>>>(tmp, cls_tg + i * 256,
                                                        cls_tbeta + i * 256, abf, HW);
            curx = abf;
        }
        {   // cls head (80 ch) -> tmp, then scatter into out[..., 5:85]
            dim3 grid(nblks, 2);
            conv3x3_wmma_kernel<<<grid, 256, 0, stream>>>(curx, wpCH, cls_hb, tmp, Bn, H, W, 80);
            copy_cls_kernel<<<(Ntot * 80 + 255) / 256, 256, 0, stream>>>(tmp, out, Ntot, HW, offs[j]);
        }

        // ---- reg tower ----
        curx = xbf;
        for (int i = 0; i < 4; ++i) {
            dim3 grid(nblks, 4);
            conv3x3_wmma_kernel<<<grid, 256, 0, stream>>>(curx, wpB + (size_t)i * 256 * 2304,
                                                          reg_tb + i * 256, tmp, Bn, H, W, 256);
            gn_relu_kernel<<<Bn * 32, 256, 0, stream>>>(tmp, reg_tg + i * 256,
                                                        reg_tbeta + i * 256, abf, HW);
            curx = abf;
        }
        {   // reg+iou head (5 ch) -> tmp, then decode boxes into out[..., 0:5]
            dim3 grid(nblks, 1);
            conv3x3_wmma_kernel<<<grid, 256, 0, stream>>>(curx, wpRI, b5, tmp, Bn, H, W, 5);
            decode_kernel<<<(Ntot + 255) / 256, 256, 0, stream>>>(tmp, scales, j, out, Ntot, HW, W,
                                                                  strides[j], asz[j], offs[j]);
        }
    }
}
